// MyVisionMamba_6262062317680
// MI455X (gfx1250) — compile-verified, matrix operand order checked by reading, not run on hardware
//
#include <hip/hip_runtime.h>
#include <hip/hip_bf16.h>
#include <math.h>
#include <stdint.h>

// ---------------- model dims ----------------
#define D_MODEL   192
#define D_INNER   384
#define D_STATE   16
#define DT_RANK   12
#define D_CONV    4
#define DEPTH     24
#define SEQ       257
#define BATCH     32
#define NPATCH    256
#define TOK       (BATCH * SEQ)      // 8224  (multiple of 16)
#define TOKP      (BATCH * NPATCH)   // 8192
#define PROJ_N    (DT_RANK + 2 * D_STATE)   // 44 (padded to 48 for GEMM)
#define PROJ_LD   48
#define PATCH_K   (3 * 16 * 16)      // 768
#define EPS_F     1e-5f

typedef unsigned short u16;
typedef __attribute__((ext_vector_type(16))) __bf16 v16bf;
typedef __attribute__((ext_vector_type(8)))  float  v8f;

// ---------------- helpers ----------------
static __device__ __forceinline__ u16 f2b(float f) {
  // round-to-nearest-even fp32 -> bf16 bits
  unsigned int x = __builtin_bit_cast(unsigned int, f);
  unsigned int r = x + 0x7FFFu + ((x >> 16) & 1u);
  return (u16)(r >> 16);
}
static __device__ __forceinline__ __bf16 b16(u16 u) {
  return __builtin_bit_cast(__bf16, u);
}
static __device__ __forceinline__ float siluf(float v) {
  return v / (1.f + __expf(-v));
}
// low 32 bits of a generic pointer to __shared__ = wave-relative LDS byte address
static __device__ __forceinline__ unsigned lds_addr32(const void* p) {
  return (unsigned)(uintptr_t)p;
}
// 16B async DMA chunk: LDS[dst] <= MEM[saddr + voff]  (tracked by ASYNCcnt)
static __device__ __forceinline__ void async_ld_b128(unsigned dst, unsigned voff,
                                                     unsigned long long saddr) {
  asm volatile("global_load_async_to_lds_b128 %0, %1, %2"
               :: "v"(dst), "v"(voff), "s"(saddr) : "memory");
}
// wait until outstanding async ops <= N (in-order completion => tile ready)
template <int N>
static __device__ __forceinline__ void wait_async() {
  asm volatile("s_wait_asynccnt %0" :: "i"(N) : "memory");
}

// K-index of fragment element e for this lane (16-bit A 16x32 layout, ISA 7.12.2;
// B 32x16 uses the mirrored mapping with N = lane&15).
static __device__ __forceinline__ int frag_k(int lane, int e) {
  int base = (lane & 16) ? 8 : 0;
  int v = e >> 1, h = e & 1;
  return base + ((v < 4) ? (v * 2 + h) : (16 + (v - 4) * 2 + h));
}

// ---------------- WMMA GEMM: C[M,N] = A[M,K] * Bw[N,K]^T ----------------
// Requirements: M % 16 == 0, K % 32 == 0, N % (16*NT) == 0.
// One wave per block; wave owns a 16 x (16*NT) output strip. A/B tiles are
// double-buffered in LDS via global_load_async_to_lds_b128 (ASYNCcnt): the
// next K-step's DMA batch is issued before waiting (threshold = batch size,
// legal because async loads complete in order), so DMA overlaps the WMMAs.
template <int NT>
__global__ void __launch_bounds__(32)
gemm_bf16_wmma(const u16* __restrict__ A, const u16* __restrict__ Bw,
               float* __restrict__ C, int M, int N, int K, int ldc) {
  constexpr int CHUNKS = 2 + 2 * NT;     // async instrs per K-step batch
  __shared__ u16 sA[2][16 * 32];
  __shared__ u16 sB[2][NT * 16 * 32];
  const int lane = threadIdx.x & 31;
  const int m0 = blockIdx.x * 16;
  const int n0 = blockIdx.y * (16 * NT);

  const unsigned ldsA[2] = {lds_addr32(sA[0]), lds_addr32(sA[1])};
  const unsigned ldsB[2] = {lds_addr32(sB[0]), lds_addr32(sB[1])};
  const unsigned long long baseA =
      (unsigned long long)(uintptr_t)(A + (size_t)m0 * K);
  const unsigned long long baseB =
      (unsigned long long)(uintptr_t)(Bw + (size_t)n0 * K);

  // issue one K-step staging batch (CHUNKS async instructions) into buffer buf
  auto stage = [&](int kk, int buf) {
#pragma unroll
    for (int j = 0; j < 2; ++j) {              // A tile: 64 x 16B chunks
      int q = lane + 32 * j;
      int row = q >> 2, c = q & 3;
      async_ld_b128(ldsA[buf] + (unsigned)(row * 64 + c * 16),
                    (unsigned)((row * K + kk + c * 8) * 2), baseA);
    }
#pragma unroll
    for (int j = 0; j < 2 * NT; ++j) {         // B strip: 64*NT x 16B chunks
      int q = lane + 32 * j;
      int row = q >> 2, c = q & 3;
      async_ld_b128(ldsB[buf] + (unsigned)(row * 64 + c * 16),
                    (unsigned)((row * K + kk + c * 8) * 2), baseB);
    }
  };

  v8f acc[NT];
#pragma unroll
  for (int t = 0; t < NT; ++t) acc[t] = (v8f){};

  stage(0, 0);
  int cur = 0;
  for (int kk = 0; kk < K; kk += 32) {
    const bool has_next = (kk + 32) < K;
    if (has_next) {
      stage(kk + 32, cur ^ 1);
      wait_async<CHUNKS>();                    // buffer `cur` landed; next in flight
    } else {
      wait_async<0>();
    }
    v16bf fa;
#pragma unroll
    for (int e = 0; e < 16; ++e)
      fa[e] = b16(sA[cur][(lane & 15) * 32 + frag_k(lane, e)]);
#pragma unroll
    for (int t = 0; t < NT; ++t) {
      v16bf fb;
#pragma unroll
      for (int e = 0; e < 16; ++e)
        fb[e] = b16(sB[cur][(t * 16 + (lane & 15)) * 32 + frag_k(lane, e)]);
      acc[t] = __builtin_amdgcn_wmma_f32_16x16x32_bf16(false, fa, false, fb,
                                                       (short)0, acc[t],
                                                       false, false);
    }
    cur ^= 1;
  }
  const int mb = m0 + ((lane & 16) ? 8 : 0);
#pragma unroll
  for (int t = 0; t < NT; ++t) {
    const int n = n0 + t * 16 + (lane & 15);
#pragma unroll
    for (int r = 0; r < 8; ++r) C[(size_t)(mb + r) * ldc + n] = acc[t][r];
  }
}

// ---------------- elementwise kernels ----------------
__global__ void convert_f32_to_bf16(const float* __restrict__ in, u16* __restrict__ out, int n) {
  int i = blockIdx.x * blockDim.x + threadIdx.x;
  if (i < n) out[i] = f2b(in[i]);
}

// x_proj weights: [DEPTH][44][384] f32 -> [DEPTH][48][384] bf16, rows 44..47 = 0
__global__ void convert_pad_xproj(const float* __restrict__ in, u16* __restrict__ out) {
  int i = blockIdx.x * blockDim.x + threadIdx.x;
  if (i >= DEPTH * PROJ_LD * D_INNER) return;
  int k = i % D_INNER;
  int n = (i / D_INNER) % PROJ_LD;
  int layer = i / (D_INNER * PROJ_LD);
  out[i] = (n < PROJ_N)
               ? f2b(in[((size_t)layer * PROJ_N + n) * D_INNER + k])
               : (u16)0;
}

// im2col for 16x16/stride16 patch conv: A2[m, k] with m=(b,patch), k=(c,py,px)
__global__ void im2col_patches(const float* __restrict__ x, u16* __restrict__ a, int total) {
  int i = blockIdx.x * blockDim.x + threadIdx.x;
  if (i >= total) return;
  int k = i % PATCH_K;
  int m = i / PATCH_K;
  int b = m / NPATCH, p = m % NPATCH;
  int c = k / 256, rem = k % 256;
  int py = rem / 16, px = rem % 16;
  int pr = p / 16, pc = p % 16;
  a[i] = f2b(x[(((size_t)b * 3 + c) * 256 + (pr * 16 + py)) * 256 + (pc * 16 + px)]);
}

// insert cls token at SEQ/2, add bias + positional embedding -> residual f32
__global__ void assemble_tokens(const float* __restrict__ feat, const float* __restrict__ patch_b,
                                const float* __restrict__ cls, const float* __restrict__ pos,
                                float* __restrict__ residual) {
  int i = blockIdx.x * blockDim.x + threadIdx.x;
  if (i >= TOK * D_MODEL) return;
  int d = i % D_MODEL;
  int l = (i / D_MODEL) % SEQ;
  int b = i / (D_MODEL * SEQ);
  float v;
  if (l == NPATCH / 2) {
    v = cls[d];
  } else {
    int p = (l < NPATCH / 2) ? l : l - 1;
    v = feat[((size_t)b * NPATCH + p) * D_MODEL + d] + patch_b[d];
  }
  residual[i] = v + pos[l * D_MODEL + d];
}

// fused residual-add + RMSNorm; warp-per-token (192 = 6 elems/lane).
// out_bf != null -> bf16 output (GEMM input); else f32 to out_f (final output).
__global__ void __launch_bounds__(256)
add_rmsnorm(float* __restrict__ residual, const float* __restrict__ hin,
            const float* __restrict__ w, u16* __restrict__ out_bf,
            float* __restrict__ out_f, int add_h) {
  int wid = threadIdx.x >> 5, lane = threadIdx.x & 31;
  int tok = blockIdx.x * 8 + wid;
  if (tok >= TOK) return;
  float v[6];
  float ss = 0.f;
#pragma unroll
  for (int j = 0; j < 6; ++j) {
    int d = lane + j * 32;
    float r = residual[(size_t)tok * D_MODEL + d];
    if (add_h) {
      r += hin[(size_t)tok * D_MODEL + d];
      residual[(size_t)tok * D_MODEL + d] = r;
    }
    v[j] = r;
    ss += r * r;
  }
#pragma unroll
  for (int m = 16; m >= 1; m >>= 1) ss += __shfl_xor(ss, m, 32);
  float sc = rsqrtf(ss / (float)D_MODEL + EPS_F);
#pragma unroll
  for (int j = 0; j < 6; ++j) {
    int d = lane + j * 32;
    float o = v[j] * sc * w[d];
    if (out_bf) out_bf[(size_t)tok * D_MODEL + d] = f2b(o);
    else        out_f[(size_t)tok * D_MODEL + d] = o;
  }
}

// causal depthwise conv + SiLU, in LOGICAL scan order (rev flips source index).
// xz = [TOK, 2*D_INNER]; xin is the first half.
__global__ void conv_silu(const float* __restrict__ xz, const float* __restrict__ cw,
                          const float* __restrict__ cb, float* __restrict__ xf,
                          u16* __restrict__ xb, int rev) {
  int i = blockIdx.x * blockDim.x + threadIdx.x;
  if (i >= TOK * D_INNER) return;
  int d = i % D_INNER;
  int t = (i / D_INNER) % SEQ;
  int b = i / (D_INNER * SEQ);
  float acc = cb[d];
#pragma unroll
  for (int k = 0; k < D_CONV; ++k) {
    int tl = t - (D_CONV - 1) + k;
    if (tl >= 0) {
      int lp = rev ? (SEQ - 1 - tl) : tl;
      acc += xz[((size_t)b * SEQ + lp) * (2 * D_INNER) + d] * cw[(size_t)d * D_CONV + k];
    }
  }
  float s = siluf(acc);
  xf[i] = s;
  xb[i] = f2b(s);
}

// dt = softplus(proj[:, :DT_RANK] @ dtw^T + dtb)  (K=12, VALU)
__global__ void dt_softplus(const float* __restrict__ proj, const float* __restrict__ dtw,
                            const float* __restrict__ dtb, float* __restrict__ dt) {
  int i = blockIdx.x * blockDim.x + threadIdx.x;
  if (i >= TOK * D_INNER) return;
  int d = i % D_INNER;
  int tok = i / D_INNER;
  float a = dtb[d];
#pragma unroll
  for (int r = 0; r < DT_RANK; ++r)
    a += proj[(size_t)tok * PROJ_LD + r] * dtw[(size_t)d * DT_RANK + r];
  dt[i] = (a > 20.f) ? a : log1pf(__expf(a));
}

// selective scan: one block per batch element, one thread per channel d.
// 16-state recurrence in registers, B_t/C_t staged through LDS per step.
// Inputs are in logical time order; output written at physical index.
__global__ void __launch_bounds__(D_INNER)
selective_scan(const float* __restrict__ dt, const float* __restrict__ proj,
               const float* __restrict__ xc, const float* __restrict__ A_log,
               const float* __restrict__ Dskip, float* __restrict__ y, int rev) {
  __shared__ float sBC[2 * D_STATE];
  int b = blockIdx.x, d = threadIdx.x;
  float a[D_STATE], h[D_STATE];
#pragma unroll
  for (int n = 0; n < D_STATE; ++n) {
    a[n] = -__expf(A_log[(size_t)d * D_STATE + n]);
    h[n] = 0.f;
  }
  float Dp = Dskip[d];
  for (int t = 0; t < SEQ; ++t) {
    size_t tok = (size_t)b * SEQ + t;
    if (d < 2 * D_STATE) sBC[d] = proj[tok * PROJ_LD + DT_RANK + d];
    __syncthreads();
    float dtv = dt[tok * D_INNER + d];
    float xv  = xc[tok * D_INNER + d];
    float dx = dtv * xv;
    float yv = 0.f;
#pragma unroll
    for (int n = 0; n < D_STATE; ++n) {
      float dA = __expf(dtv * a[n]);            // ZOH discretization
      h[n] = h[n] * dA + dx * sBC[n];
      yv += h[n] * sBC[D_STATE + n];
    }
    yv += Dp * xv;
    int lp = rev ? (SEQ - 1 - t) : t;
    y[((size_t)b * SEQ + lp) * D_INNER + d] = yv;
    __syncthreads();
  }
}

// y = 0.5*(y_f + y_r) * silu(z)  -> bf16 for out_proj GEMM
__global__ void combine_y(const float* __restrict__ yf, const float* __restrict__ yr,
                          const float* __restrict__ xz, u16* __restrict__ yb) {
  int i = blockIdx.x * blockDim.x + threadIdx.x;
  if (i >= TOK * D_INNER) return;
  int d = i % D_INNER;
  int tok = i / D_INNER;
  float z = xz[(size_t)tok * (2 * D_INNER) + D_INNER + d];
  yb[i] = f2b(0.5f * (yf[i] + yr[i]) * siluf(z));
}

// ---------------- host orchestration ----------------
extern "C" void kernel_launch(void* const* d_in, const int* in_sizes, int n_in,
                              void* d_out, int out_size, void* d_ws, size_t ws_size,
                              hipStream_t stream) {
  (void)in_sizes; (void)n_in; (void)out_size; (void)ws_size;
  const float* x         = (const float*)d_in[0];
  const float* patch_w   = (const float*)d_in[1];
  const float* patch_b   = (const float*)d_in[2];
  const float* cls_token = (const float*)d_in[3];
  const float* pos_embed = (const float*)d_in[4];
  const float* norm_w    = (const float*)d_in[5];
  const float* in_proj_w = (const float*)d_in[6];
  const float* conv_w[2]   = {(const float*)d_in[7],  (const float*)d_in[14]};
  const float* conv_bias[2]= {(const float*)d_in[8],  (const float*)d_in[15]};
  const float* xproj_w[2]  = {(const float*)d_in[9],  (const float*)d_in[16]};
  const float* dtproj_w[2] = {(const float*)d_in[10], (const float*)d_in[17]};
  const float* dt_bias[2]  = {(const float*)d_in[11], (const float*)d_in[18]};
  const float* A_log[2]    = {(const float*)d_in[12], (const float*)d_in[19]};
  const float* Dskip[2]    = {(const float*)d_in[13], (const float*)d_in[20]};
  const float* out_proj_w  = (const float*)d_in[21];
  const float* norm_f_w    = (const float*)d_in[22];
  float* out = (float*)d_out;

  // workspace carve-out
  char* wp = (char*)d_ws;
  auto alloc = [&](size_t bytes) -> void* {
    void* p = (void*)wp;
    wp += (bytes + 255) & ~(size_t)255;
    return p;
  };
  float* residual = (float*)alloc((size_t)TOK * D_MODEL * 4);
  float* hbuf     = (float*)alloc((size_t)TOK * D_MODEL * 4);
  u16*   hn_bf    = (u16*)  alloc((size_t)TOK * D_MODEL * 2);
  float* xz       = (float*)alloc((size_t)TOK * 2 * D_INNER * 4);
  float* xcf[2]; u16* xcb[2]; float* projb[2]; float* dtb_buf[2]; float* yscan[2];
  for (int dir = 0; dir < 2; ++dir) {
    xcf[dir]     = (float*)alloc((size_t)TOK * D_INNER * 4);
    xcb[dir]     = (u16*)  alloc((size_t)TOK * D_INNER * 2);
    projb[dir]   = (float*)alloc((size_t)TOK * PROJ_LD * 4);
    dtb_buf[dir] = (float*)alloc((size_t)TOK * D_INNER * 4);
    yscan[dir]   = (float*)alloc((size_t)TOK * D_INNER * 4);
  }
  u16*   ycomb  = (u16*)  alloc((size_t)TOK * D_INNER * 2);
  u16*   a2col  = (u16*)  alloc((size_t)TOKP * PATCH_K * 2);
  float* feat   = (float*)alloc((size_t)TOKP * D_MODEL * 4);
  u16*   w_patch= (u16*)  alloc((size_t)D_MODEL * PATCH_K * 2);
  u16*   w_in   = (u16*)  alloc((size_t)DEPTH * 2 * D_INNER * D_MODEL * 2);
  u16*   w_x[2] = {(u16*)alloc((size_t)DEPTH * PROJ_LD * D_INNER * 2),
                   (u16*)alloc((size_t)DEPTH * PROJ_LD * D_INNER * 2)};
  u16*   w_out  = (u16*)  alloc((size_t)DEPTH * D_MODEL * D_INNER * 2);

  auto cvt = [&](const float* src, u16* dst, int cnt) {
    convert_f32_to_bf16<<<(cnt + 255) / 256, 256, 0, stream>>>(src, dst, cnt);
  };
  // NT=4 strip (N % 64 == 0)
  auto gemm4 = [&](const u16* A, const u16* Bw, float* C, int M, int N, int K, int ldc) {
    dim3 grid(M / 16, N / 64);
    gemm_bf16_wmma<4><<<grid, 32, 0, stream>>>(A, Bw, C, M, N, K, ldc);
  };
  // NT=3 strip (N % 48 == 0) — used for the padded x_proj GEMM
  auto gemm3 = [&](const u16* A, const u16* Bw, float* C, int M, int N, int K, int ldc) {
    dim3 grid(M / 16, N / 48);
    gemm_bf16_wmma<3><<<grid, 32, 0, stream>>>(A, Bw, C, M, N, K, ldc);
  };

  // ---- weight conversion (per launch: deterministic, no caching) ----
  cvt(patch_w, w_patch, D_MODEL * PATCH_K);
  cvt(in_proj_w, w_in, DEPTH * 2 * D_INNER * D_MODEL);
  cvt(out_proj_w, w_out, DEPTH * D_MODEL * D_INNER);
  {
    int tot = DEPTH * PROJ_LD * D_INNER;
    convert_pad_xproj<<<(tot + 255) / 256, 256, 0, stream>>>(xproj_w[0], w_x[0]);
    convert_pad_xproj<<<(tot + 255) / 256, 256, 0, stream>>>(xproj_w[1], w_x[1]);
  }

  // ---- patch embedding ----
  {
    int tot = TOKP * PATCH_K;
    im2col_patches<<<(tot + 255) / 256, 256, 0, stream>>>(x, a2col, tot);
    gemm4(a2col, w_patch, feat, TOKP, D_MODEL, PATCH_K, D_MODEL);
    int tot2 = TOK * D_MODEL;
    assemble_tokens<<<(tot2 + 255) / 256, 256, 0, stream>>>(feat, patch_b, cls_token,
                                                            pos_embed, residual);
  }

  const int NBLK_NORM = TOK / 8;                 // 1028
  const int TOT_DI = TOK * D_INNER;
  const int GRID_DI = (TOT_DI + 255) / 256;

  // ---- 24 mamba layers ----
  for (int layer = 0; layer < DEPTH; ++layer) {
    add_rmsnorm<<<NBLK_NORM, 256, 0, stream>>>(residual, hbuf, norm_w + layer * D_MODEL,
                                               hn_bf, nullptr, layer > 0);
    gemm4(hn_bf, w_in + (size_t)layer * 2 * D_INNER * D_MODEL, xz,
          TOK, 2 * D_INNER, D_MODEL, 2 * D_INNER);

    for (int dir = 0; dir < 2; ++dir) {
      conv_silu<<<GRID_DI, 256, 0, stream>>>(
          xz, conv_w[dir] + (size_t)layer * D_INNER * D_CONV,
          conv_bias[dir] + (size_t)layer * D_INNER, xcf[dir], xcb[dir], dir);
      gemm3(xcb[dir], w_x[dir] + (size_t)layer * PROJ_LD * D_INNER, projb[dir],
            TOK, PROJ_LD, D_INNER, PROJ_LD);
      dt_softplus<<<GRID_DI, 256, 0, stream>>>(
          projb[dir], dtproj_w[dir] + (size_t)layer * D_INNER * DT_RANK,
          dt_bias[dir] + (size_t)layer * D_INNER, dtb_buf[dir]);
      selective_scan<<<BATCH, D_INNER, 0, stream>>>(
          dtb_buf[dir], projb[dir], xcf[dir],
          A_log[dir] + (size_t)layer * D_INNER * D_STATE,
          Dskip[dir] + (size_t)layer * D_INNER, yscan[dir], dir);
    }

    combine_y<<<GRID_DI, 256, 0, stream>>>(yscan[0], yscan[1], xz, ycomb);
    gemm4(ycomb, w_out + (size_t)layer * D_MODEL * D_INNER, hbuf,
          TOK, D_MODEL, D_INNER, D_MODEL);
  }

  // ---- final fused add + RMSNorm -> f32 output ----
  add_rmsnorm<<<NBLK_NORM, 256, 0, stream>>>(residual, hbuf, norm_f_w,
                                             nullptr, out, 1);
}
